// SDCN_3607772529207
// MI455X (gfx1250) — matrix operand order repaired and checked by reading, not compiled, not run on hardware
//
#include <hip/hip_runtime.h>
#include <hip/hip_bf16.h>
#include <stdint.h>

// ---------------------------------------------------------------------------
// SDCN forward for MI455X (gfx1250, wave32).
//   GEMM: bf16 operands, v_wmma_f32_16x16x32_bf16, double-buffered LDS.
//         A-tile staging: TDM tensor_load_to_lds (mode 2, with native LDS
//         padding + OOB zero-fill) or global_load_async_to_lds_b64 (mode 1)
//         or clamped scalar loads (mode 0 / tails).
//   SpMM: edge-list f32 global atomics (L2-resident RMW).
// ---------------------------------------------------------------------------

typedef unsigned short ushort_t;
typedef __attribute__((ext_vector_type(16))) __bf16        v16bf;
typedef __attribute__((ext_vector_type(8)))  float         v8f;
typedef __attribute__((ext_vector_type(4)))  unsigned int  v4u;
typedef __attribute__((ext_vector_type(8)))  int           v8i_t;
typedef __attribute__((ext_vector_type(4)))  int           v4i_t;

static constexpr int BM  = 128;      // block rows
static constexpr int BN  = 64;       // block cols
static constexpr int BK  = 32;       // K tile
static constexpr int LDA = 40;       // As stride (bf16): 80B rows (64B + 16B pad)
static constexpr int LDB = 34;       // Bs stride (bf16), transposed [n][k]

__device__ __forceinline__ ushort_t f32_to_bf16(float f) {
    unsigned u = __float_as_uint(f);
    u += 0x7FFFu + ((u >> 16) & 1u);                 // round-to-nearest-even
    return (ushort_t)(u >> 16);
}
__device__ __forceinline__ float bf16_to_f32(ushort_t u) {
    return __uint_as_float((unsigned)u << 16);
}

// ---------------------------------------------------------------------------
// A-tile staging paths
// ---------------------------------------------------------------------------

// TDM: one descriptor moves the whole 128x32 bf16 tile, padding each 64B row
// with 16B in LDS (-> 80B = LDA rows) and zero-filling past tensor extents.
__device__ __forceinline__ void tdm_load_A(const ushort_t* gsrc, uint32_t lds_addr,
                                           int remK, int remM, int Kdim)
{
    const uint64_t ga = (uint64_t)(uintptr_t)gsrc;
    v4u g0;
    g0.x = 1u;                                    // count=1, user descriptor
    g0.y = lds_addr;                              // LDS byte address
    g0.z = (uint32_t)ga;                          // global_addr[31:0]
    g0.w = (uint32_t)((ga >> 32) & 0x01FFFFFFull) // global_addr[56:32]
         | (2u << 30);                            // type = 2 ("image")

    v8i_t g1;
    // data_size=1 (2B) | pad_enable | pad_interval=3 (16 DW) | pad_amount=3 (4 DW)
    g1[0] = (1 << 16) | (1 << 20) | (3 << 22) | (3 << 25);
    g1[1] = (int)(((uint32_t)remK & 0xFFFFu) << 16);              // tensor_dim0[15:0]
    g1[2] = (int)((((uint32_t)remK >> 16) & 0xFFFFu)              // tensor_dim0[31:16]
                | (((uint32_t)remM & 0xFFFFu) << 16));            // tensor_dim1[15:0]
    g1[3] = (int)((((uint32_t)remM >> 16) & 0xFFFFu)              // tensor_dim1[31:16]
                | ((uint32_t)BK << 16));                          // tile_dim0 = 32
    g1[4] = BM;                                                   // tile_dim1=128, tile_dim2=0
    g1[5] = Kdim;                                                 // tensor_dim0_stride[31:0]
    g1[6] = 0;                                                    // stride0 hi / stride1 lo
    g1[7] = 0;
    const v4i_t z4 = {0, 0, 0, 0};
    const v8i_t z8 = {0, 0, 0, 0, 0, 0, 0, 0};
    // clang-23 / therock form: (g0, g1, grp2, grp3, extra, cpol)
    __builtin_amdgcn_tensor_load_to_lds(g0, g1, z4, z4, z8, 0);
}

__device__ __forceinline__ void stageA_async(const ushort_t* __restrict__ Ab,
                                             uint32_t ldsBase, int blockRow,
                                             int M, int Kdim, int k0, int tid)
{
    // 128x32 bf16 tile = 1024 x 8B chunks; 4 async b64 copies per thread.
    #pragma unroll
    for (int i = 0; i < 4; ++i) {
        const int ch  = tid + i * 256;
        const int row = ch >> 3;
        const int cc  = ch & 7;
        int grow = blockRow + row;
        if (grow >= M) grow = M - 1;                 // clamp; masked at store
        const ushort_t* g = Ab + (size_t)grow * (size_t)Kdim + (size_t)(k0 + cc * 4);
        const uint32_t  lo = ldsBase + (uint32_t)(row * LDA + cc * 4) * 2u;
        asm volatile("global_load_async_to_lds_b64 %0, %1, off"
                     :: "v"(lo), "v"(g) : "memory");
    }
}

__device__ __forceinline__ void stageA_scalar(const ushort_t* __restrict__ Ab,
                                              ushort_t* __restrict__ As_,
                                              int blockRow, int M, int Kdim,
                                              int k0, int tid)
{
    const int c  = tid & 31;
    const int r0 = tid >> 5;
    #pragma unroll
    for (int it = 0; it < 16; ++it) {
        const int row  = r0 + it * 8;
        const int grow = blockRow + row;
        const int gk   = k0 + c;
        const int rc   = (grow < M) ? grow : (M - 1);
        const int kc   = (gk < Kdim) ? gk : (Kdim - 1);
        ushort_t v = Ab[(size_t)rc * (size_t)Kdim + kc];
        if (grow >= M || gk >= Kdim) v = 0;
        As_[row * LDA + c] = v;
    }
}

__device__ __forceinline__ void stageA(int mode, const ushort_t* __restrict__ Ab,
                                       ushort_t* __restrict__ AsBuf,
                                       int blockRow, int M, int Kdim, int k0,
                                       int tid, int wave)
{
    const bool full = (k0 + BK) <= Kdim;
    if (mode == 2 && full) {
        if (wave == 0)
            tdm_load_A(Ab + (size_t)blockRow * (size_t)Kdim + k0,
                       (uint32_t)(uintptr_t)AsBuf, Kdim - k0, M - blockRow, Kdim);
    } else if (mode == 1 && full) {
        stageA_async(Ab, (uint32_t)(uintptr_t)AsBuf, blockRow, M, Kdim, k0, tid);
    } else {
        stageA_scalar(Ab, AsBuf, blockRow, M, Kdim, k0, tid);
    }
}

__device__ __forceinline__ void stageB(const ushort_t* __restrict__ Wb,
                                       ushort_t* __restrict__ Bs_,
                                       int blockCol, int Kdim, int Nout,
                                       int k0, int tid)
{
    const int c  = tid & 63;
    const int r0 = tid >> 6;
    const int gc = blockCol + c;
    const int cc = (gc < Nout) ? gc : (Nout - 1);
    #pragma unroll
    for (int it = 0; it < 8; ++it) {
        const int kk = r0 + it * 4;
        const int gk = k0 + kk;
        const int kc = (gk < Kdim) ? gk : (Kdim - 1);
        ushort_t v = Wb[(size_t)kc * (size_t)Nout + cc];
        if (gk >= Kdim || gc >= Nout) v = 0;
        Bs_[c * LDB + kk] = v;                       // transposed [n][k]
    }
}

// ---------------------------------------------------------------------------
// GEMM: out = act(A[M,K](bf16) @ W[K,N](bf16) + bias), dual f32/bf16 outputs.
// ---------------------------------------------------------------------------
__global__ __launch_bounds__(256)
void gemm_bf16_wmma(const ushort_t* __restrict__ Ab,
                    const ushort_t* __restrict__ Wb,
                    const float* __restrict__ bias,
                    float* __restrict__ outF, ushort_t* __restrict__ outB,
                    int M, int Kdim, int Nout, int relu, int mode)
{
    __shared__ ushort_t As[2][BM * LDA];             // 2 x 10240B
    __shared__ ushort_t Bs[2][BN * LDB];             // 2 x  4352B

    const int tid   = threadIdx.x;
    const int lane  = tid & 31;
    const int wave  = tid >> 5;
    const int waveM = wave >> 1;                     // 4 waves along M
    const int waveN = wave & 1;                      // 2 waves along N
    const int half  = lane >> 4;
    const int l16   = lane & 15;

    const int blockRow = blockIdx.y * BM;
    const int blockCol = blockIdx.x * BN;
    const int nk = (Kdim + BK - 1) / BK;

    union Acc  { v8f  v; float    f[8]; };
    union Frag { v16bf v; uint32_t u[8]; };

    Acc acc[2][2];
    #pragma unroll
    for (int i = 0; i < 2; ++i)
        #pragma unroll
        for (int j = 0; j < 2; ++j)
            #pragma unroll
            for (int r = 0; r < 8; ++r) acc[i][j].f[r] = 0.0f;

    // ---- prologue: stage tile 0 into buffer 0 ----------------------------
    stageA(mode, Ab, &As[0][0], blockRow, M, Kdim, 0, tid, wave);
    stageB(Wb, &Bs[0][0], blockCol, Kdim, Nout, 0, tid);
    asm volatile("s_wait_asynccnt 0x0" ::: "memory");
    __builtin_amdgcn_s_wait_tensorcnt(0);
    __syncthreads();

    for (int kt = 0; kt < nk; ++kt) {
        const int cur = kt & 1;
        const int nxt = cur ^ 1;

        // ---- stage tile kt+1 into the other buffer (overlaps compute) ----
        if (kt + 1 < nk) {
            const int k0n = (kt + 1) * BK;
            stageA(mode, Ab, &As[nxt][0], blockRow, M, Kdim, k0n, tid, wave);
            stageB(Wb, &Bs[nxt][0], blockCol, Kdim, Nout, k0n, tid);
        }

        // ---- fragment loads (CDNA5 ISA VGPR layouts) ---------------------
        Frag a[2], b[2];
        #pragma unroll
        for (int i = 0; i < 2; ++i) {
            const int row = waveM * 32 + i * 16 + l16;          // A lane = M
            const uint32_t* base = (const uint32_t*)&As[cur][row * LDA];
            #pragma unroll
            for (int v = 0; v < 8; ++v) {
                const int j  = 2 * v;
                const int kk = j + half * 8 + ((j >= 8) ? 8 : 0);
                a[i].u[v] = base[kk >> 1];
            }
        }
        #pragma unroll
        for (int jj = 0; jj < 2; ++jj) {
            const int n = waveN * 32 + jj * 16 + l16;           // B lane = N
            const uint32_t* base = (const uint32_t*)&Bs[cur][n * LDB];
            #pragma unroll
            for (int v = 0; v < 8; ++v) {
                const int kk = 2 * v + 8 * half + ((v >= 4) ? 8 : 0);
                b[jj].u[v] = base[kk >> 1];
            }
        }

        // ---- 2x2 WMMA, 32x32 per wave (EXEC all-1s) ----------------------
        #pragma unroll
        for (int i = 0; i < 2; ++i)
            #pragma unroll
            for (int jj = 0; jj < 2; ++jj)
                acc[i][jj].v = __builtin_amdgcn_wmma_f32_16x16x32_bf16(
                    false, a[i].v, false, b[jj].v,
                    (short)0, acc[i][jj].v, false, false);

        asm volatile("s_wait_asynccnt 0x0" ::: "memory");
        __builtin_amdgcn_s_wait_tensorcnt(0);
        __syncthreads();
    }

    // ---- epilogue: +bias, relu, dual f32 / bf16 stores -------------------
    #pragma unroll
    for (int jj = 0; jj < 2; ++jj) {
        const int n = blockCol + waveN * 32 + jj * 16 + l16;
        const float bval = (bias && n < Nout) ? bias[n] : 0.0f;
        #pragma unroll
        for (int i = 0; i < 2; ++i) {
            #pragma unroll
            for (int r = 0; r < 8; ++r) {
                const int m = blockRow + waveM * 32 + i * 16 + r + 8 * half;
                if (m < M && n < Nout) {
                    float v = acc[i][jj].f[r] + bval;
                    if (relu) v = fmaxf(v, 0.0f);
                    const size_t o = (size_t)m * (size_t)Nout + n;
                    if (outF) outF[o] = v;
                    if (outB) outB[o] = f32_to_bf16(v);
                }
            }
        }
    }
}

// ---------------------------------------------------------------------------
__global__ __launch_bounds__(256)
void cvt_bf16(const float* __restrict__ s, ushort_t* __restrict__ d, long long n) {
    long long i = (long long)blockIdx.x * blockDim.x + threadIdx.x;
    const long long st = (long long)gridDim.x * blockDim.x;
    for (; i < n; i += st) d[i] = f32_to_bf16(s[i]);
}

__global__ __launch_bounds__(256)
void fill_zero(float* __restrict__ p, long long n) {
    long long i = (long long)blockIdx.x * blockDim.x + threadIdx.x;
    const long long st = (long long)gridDim.x * blockDim.x;
    for (; i < n; i += st) p[i] = 0.0f;
}

__global__ __launch_bounds__(256)
void relu_bf16(float* __restrict__ p, ushort_t* __restrict__ pb, long long n) {
    long long i = (long long)blockIdx.x * blockDim.x + threadIdx.x;
    const long long st = (long long)gridDim.x * blockDim.x;
    for (; i < n; i += st) {
        const float v = fmaxf(p[i], 0.0f);
        p[i] = v;
        pb[i] = f32_to_bf16(v);
    }
}

__global__ __launch_bounds__(256)
void blend_bf16(const ushort_t* __restrict__ a, const ushort_t* __restrict__ b,
                ushort_t* __restrict__ o, float alpha, float beta, long long n) {
    long long i = (long long)blockIdx.x * blockDim.x + threadIdx.x;
    const long long st = (long long)gridDim.x * blockDim.x;
    for (; i < n; i += st)
        o[i] = f32_to_bf16(alpha * bf16_to_f32(a[i]) + beta * bf16_to_f32(b[i]));
}

template<int F>
__global__ __launch_bounds__(256)
void spmm_atomic_t(const float* __restrict__ h, const float* __restrict__ ew,
                   const int* __restrict__ src, const int* __restrict__ dst,
                   float* __restrict__ out, long long total)
{
    const long long idx = (long long)blockIdx.x * blockDim.x + threadIdx.x;
    if (idx >= total) return;
    const int e = (int)(idx / F);
    const int f = (int)(idx % F);
    const int s = src[e];
    const int d = dst[e];
    const float v = ew[e] * h[(size_t)s * F + f];
    atomicAdd(&out[(size_t)d * F + f], v);
}

__global__ __launch_bounds__(256)
void softmax_small(const float* __restrict__ in, float* __restrict__ out,
                   int n, int F)
{
    const int r = blockIdx.x * blockDim.x + threadIdx.x;
    if (r >= n) return;
    float m = -3.0e38f;
    for (int i = 0; i < F; ++i) m = fmaxf(m, in[(size_t)r * F + i]);
    float e[16];
    float s = 0.0f;
    for (int i = 0; i < F; ++i) { e[i] = __expf(in[(size_t)r * F + i] - m); s += e[i]; }
    const float inv = 1.0f / s;
    for (int i = 0; i < F; ++i) out[(size_t)r * F + i] = e[i] * inv;
}

__global__ __launch_bounds__(256)
void cluster_q(const float* __restrict__ z, const float* __restrict__ cl,
               float* __restrict__ q, int n, int K, int NZ)
{
    __shared__ float c[256];
    if (threadIdx.x < K * NZ) c[threadIdx.x] = cl[threadIdx.x];
    __syncthreads();
    const int r = blockIdx.x * blockDim.x + threadIdx.x;
    if (r >= n) return;
    float zz[16];
    for (int i = 0; i < NZ; ++i) zz[i] = z[(size_t)r * NZ + i];
    float qv[16];
    float s = 0.0f;
    for (int k = 0; k < K; ++k) {
        float d = 0.0f;
        for (int i = 0; i < NZ; ++i) { const float t = zz[i] - c[k * NZ + i]; d += t * t; }
        const float val = 1.0f / (1.0f + d);
        qv[k] = val; s += val;
    }
    const float inv = 1.0f / s;
    for (int k = 0; k < K; ++k) q[(size_t)r * K + k] = qv[k] * inv;
}

// ---------------------------------------------------------------------------
static inline int pick_mode(int Kd) {
    if (Kd % 4 != 0)  return 0;    // unaligned rows -> scalar
    if (Kd >= 1000)   return 2;    // TDM tensor_load_to_lds
    return 1;                      // global_load_async_to_lds_b64
}

static inline void launch_gemm(const ushort_t* Ab, const ushort_t* Wb,
                               const float* bias, float* outF, ushort_t* outB,
                               int M, int Kd, int Nout, int relu, hipStream_t s)
{
    dim3 grid((Nout + BN - 1) / BN, (M + BM - 1) / BM);
    gemm_bf16_wmma<<<grid, 256, 0, s>>>(Ab, Wb, bias, outF, outB,
                                        M, Kd, Nout, relu, pick_mode(Kd));
}

static inline void launch_spmm(const float* h, const float* ew,
                               const int* src, const int* dst, float* out,
                               int N, int F, long long E, hipStream_t s)
{
    fill_zero<<<2048, 256, 0, s>>>(out, (long long)N * F);
    const long long total = E * (long long)F;
    const unsigned blocks = (unsigned)((total + 255) / 256);
    if (F == 500)       spmm_atomic_t<500> <<<blocks, 256, 0, s>>>(h, ew, src, dst, out, total);
    else if (F == 2000) spmm_atomic_t<2000><<<blocks, 256, 0, s>>>(h, ew, src, dst, out, total);
    else                spmm_atomic_t<10>  <<<blocks, 256, 0, s>>>(h, ew, src, dst, out, total);
}

extern "C" void kernel_launch(void* const* d_in, const int* in_sizes, int n_in,
                              void* d_out, int out_size, void* d_ws, size_t ws_size,
                              hipStream_t stream)
{
    const int NIN = 1000, E1 = 500, E2 = 500, E3 = 2000, NZ = 10;
    const int D1 = 2000, D2 = 500, D3 = 500, K = 10;
    const float SG = 0.5f;
    const int   N  = in_sizes[0] / NIN;
    const long long E = in_sizes[1];

    const float* x      = (const float*)d_in[0];
    const float* ew     = (const float*)d_in[1];
    const float* W_f[13] = { (const float*)d_in[2],  (const float*)d_in[4],
                             (const float*)d_in[6],  (const float*)d_in[8],
                             (const float*)d_in[10], (const float*)d_in[12],
                             (const float*)d_in[14], (const float*)d_in[16],
                             (const float*)d_in[18], (const float*)d_in[19],
                             (const float*)d_in[20], (const float*)d_in[21],
                             (const float*)d_in[22] };
    const long long W_n[13] = { (long long)NIN*E1, (long long)E1*E2, (long long)E2*E3,
                                (long long)E3*NZ,  (long long)NZ*D1, (long long)D1*D2,
                                (long long)D2*D3,  (long long)D3*NIN,
                                (long long)NIN*E1, (long long)E1*E2, (long long)E2*E3,
                                (long long)E3*NZ,  (long long)NZ*K };
    const float* enc1_b = (const float*)d_in[3];
    const float* enc2_b = (const float*)d_in[5];
    const float* enc3_b = (const float*)d_in[7];
    const float* zl_b   = (const float*)d_in[9];
    const float* dec1_b = (const float*)d_in[11];
    const float* dec2_b = (const float*)d_in[13];
    const float* dec3_b = (const float*)d_in[15];
    const float* xbar_b = (const float*)d_in[17];
    const float* clus   = (const float*)d_in[23];
    const int*   esrc   = (const int*)d_in[24];
    const int*   edst   = (const int*)d_in[25];

    float* out  = (float*)d_out;
    float* xbar = out;
    float* qout = xbar + (size_t)N * NIN;
    float* pred = qout + (size_t)N * K;
    float* zout = pred + (size_t)N * K;

    // ---- workspace carve-up (16B aligned) --------------------------------
    char*  base = (char*)d_ws;
    size_t off  = 0;
    auto alloc = [&](size_t bytes) -> void* {
        void* p = base + off; off += (bytes + 15) & ~(size_t)15; return p;
    };
    float* preF = (float*)alloc((size_t)N * 2000 * 4);
    float* hA   = (float*)alloc((size_t)N * 2000 * 4);
    float* hB   = (float*)alloc((size_t)N * 500  * 4);
    float* sm1  = (float*)alloc((size_t)N * 16   * 4);
    float* sm2  = (float*)alloc((size_t)N * 16   * 4);
    ushort_t* xb    = (ushort_t*)alloc((size_t)N * NIN * 2);
    ushort_t* tra1b = (ushort_t*)alloc((size_t)N * E1  * 2);
    ushort_t* tra2b = (ushort_t*)alloc((size_t)N * E2  * 2);
    ushort_t* tra3b = (ushort_t*)alloc((size_t)N * E3  * 2);
    ushort_t* zb    = (ushort_t*)alloc((size_t)N * 16  * 2);
    ushort_t* dAb   = (ushort_t*)alloc((size_t)N * 2000 * 2);  // d1 -> h3b
    ushort_t* dBb   = (ushort_t*)alloc((size_t)N * 500  * 2);  // d2 -> h2b
    ushort_t* dCb   = (ushort_t*)alloc((size_t)N * 500  * 2);  // d3 -> h1b
    ushort_t* blb   = (ushort_t*)alloc((size_t)N * 2000 * 2);  // blend buffer
    ushort_t* h4b   = (ushort_t*)alloc((size_t)N * 16   * 2);
    ushort_t* Wb[13];
    for (int i = 0; i < 13; ++i) Wb[i] = (ushort_t*)alloc((size_t)W_n[i] * 2);

    // ---- convert operands to bf16 ---------------------------------------
    cvt_bf16<<<2048, 256, 0, stream>>>(x, xb, (long long)N * NIN);
    for (int i = 0; i < 13; ++i)
        cvt_bf16<<<256, 256, 0, stream>>>(W_f[i], Wb[i], W_n[i]);

    // ---- AE --------------------------------------------------------------
    launch_gemm(xb,    Wb[0], enc1_b, nullptr, tra1b, N, NIN, E1,  1, stream);
    launch_gemm(tra1b, Wb[1], enc2_b, nullptr, tra2b, N, E1,  E2,  1, stream);
    launch_gemm(tra2b, Wb[2], enc3_b, nullptr, tra3b, N, E2,  E3,  1, stream);
    launch_gemm(tra3b, Wb[3], zl_b,   zout,    zb,    N, E3,  NZ,  0, stream);
    launch_gemm(zb,    Wb[4], dec1_b, nullptr, dAb,   N, NZ,  D1,  1, stream);
    launch_gemm(dAb,   Wb[5], dec2_b, nullptr, dBb,   N, D1,  D2,  1, stream);
    launch_gemm(dBb,   Wb[6], dec3_b, nullptr, dCb,   N, D2,  D3,  1, stream);
    launch_gemm(dCb,   Wb[7], xbar_b, xbar,    nullptr, N, D3, NIN, 0, stream);

    // ---- GCN with skip blending -----------------------------------------
    ushort_t* h1b = dCb;
    ushort_t* h2b = dBb;
    ushort_t* h3b = dAb;

    launch_gemm(xb, Wb[8], nullptr, preF, nullptr, N, NIN, E1, 0, stream);
    launch_spmm(preF, ew, esrc, edst, hA, N, E1, E, stream);
    relu_bf16<<<2048, 256, 0, stream>>>(hA, h1b, (long long)N * E1);       // h1

    blend_bf16<<<2048, 256, 0, stream>>>(h1b, tra1b, blb, 1.f - SG, SG, (long long)N * E1);
    launch_gemm(blb, Wb[9], nullptr, preF, nullptr, N, E1, E2, 0, stream);
    launch_spmm(preF, ew, esrc, edst, hB, N, E2, E, stream);
    relu_bf16<<<2048, 256, 0, stream>>>(hB, h2b, (long long)N * E2);       // h2

    blend_bf16<<<2048, 256, 0, stream>>>(h2b, tra2b, blb, 1.f - SG, SG, (long long)N * E2);
    launch_gemm(blb, Wb[10], nullptr, preF, nullptr, N, E2, E3, 0, stream);
    launch_spmm(preF, ew, esrc, edst, hA, N, E3, E, stream);
    relu_bf16<<<2048, 256, 0, stream>>>(hA, h3b, (long long)N * E3);       // h3

    blend_bf16<<<2048, 256, 0, stream>>>(h3b, tra3b, blb, 1.f - SG, SG, (long long)N * E3);
    launch_gemm(blb, Wb[11], nullptr, sm1, nullptr, N, E3, NZ, 0, stream);
    launch_spmm(sm1, ew, esrc, edst, sm2, N, NZ, E, stream);
    relu_bf16<<<2048, 256, 0, stream>>>(sm2, h4b, (long long)N * NZ);      // h4

    blend_bf16<<<2048, 256, 0, stream>>>(h4b, zb, blb, 1.f - SG, SG, (long long)N * NZ);
    launch_gemm(blb, Wb[12], nullptr, sm1, nullptr, N, NZ, K, 0, stream);
    launch_spmm(sm1, ew, esrc, edst, sm2, N, K, E, stream);                // h5

    softmax_small<<<(N + 255) / 256, 256, 0, stream>>>(sm2, pred, N, K);
    cluster_q<<<(N + 255) / 256, 256, 0, stream>>>(zout, clus, qout, N, K, NZ);

    (void)n_in; (void)out_size; (void)ws_size;
}
